// MultiheadAttention_15882789061118
// MI455X (gfx1250) — matrix operand order repaired and checked by reading, not compiled, not run on hardware
//
#include <hip/hip_runtime.h>
#include <hip/hip_bf16.h>

// MI455X / gfx1250, wave32, WMMA bf16 path.
// Fused MHA: B=4, Sq=St=2048, E=512, H=8, hd=64.

typedef __attribute__((ext_vector_type(16))) __bf16 bf16x16;
typedef __attribute__((ext_vector_type(8)))  __bf16 bf16x8;
typedef __attribute__((ext_vector_type(8)))  float   f32x8;
typedef __attribute__((ext_vector_type(4)))  unsigned int u32x4;
typedef __attribute__((ext_vector_type(4)))  int  i32x4;
typedef __attribute__((ext_vector_type(8)))  int  i32x8;

#define BATCH 4
#define NHEAD 8
#define SEQ   2048
#define EMB   512
#define HD    64

__device__ __forceinline__ bf16x16 cat8(bf16x8 lo, bf16x8 hi) {
  return __builtin_shufflevector(lo, hi, 0,1,2,3,4,5,6,7,8,9,10,11,12,13,14,15);
}

__device__ __forceinline__ bf16x8 cvt8(const float* p) {
  float4 a = *(const float4*)p;
  float4 b = *(const float4*)(p + 4);
  bf16x8 r;
  r[0] = (__bf16)a.x; r[1] = (__bf16)a.y; r[2] = (__bf16)a.z; r[3] = (__bf16)a.w;
  r[4] = (__bf16)b.x; r[5] = (__bf16)b.y; r[6] = (__bf16)b.z; r[7] = (__bf16)b.w;
  return r;
}

__device__ __forceinline__ f32x8 wmma_bf16(bf16x16 a, bf16x16 b, f32x8 c) {
  // D = A(16x32 bf16) * B(32x16 bf16) + C(16x16 f32)
  return __builtin_amdgcn_wmma_f32_16x16x32_bf16(false, a, false, b, (short)0, c,
                                                 false, false);
}

// ---------------------------------------------------------------------------
// Tensor Data Mover helpers (gfx1250 TDM).  D# packed per CDNA5 ISA §8:
//  group0: [1:0]=count=1, [63:32]=lds_addr, [120:64]=global_addr, [127:126]=2
//  group1: [17:16]=data_size(1 => 2B), [79:48]=tensor_dim0, [111:80]=tensor_dim1,
//          [127:112]=tile_dim0, [143:128]=tile_dim1, [207:160]=dim0_stride
// ---------------------------------------------------------------------------
#if __has_builtin(__builtin_amdgcn_tensor_load_to_lds)
#define HAS_TDM 1
#if defined(__clang_major__) && (__clang_major__ >= 23)
#define TDM_ISSUE(g0, g1) \
  __builtin_amdgcn_tensor_load_to_lds(g0, g1, (i32x4)0, (i32x4)0, (i32x8)0, 0)
#else
#define TDM_ISSUE(g0, g1) \
  __builtin_amdgcn_tensor_load_to_lds(g0, g1, (i32x4)0, (i32x4)0, 0)
#endif

__device__ __forceinline__ void tdm_load_2d(unsigned lds_off, const void* gptr,
                                            unsigned tensor_d0, unsigned tensor_d1,
                                            unsigned tile_d0, unsigned tile_d1,
                                            unsigned stride0_elts) {
  unsigned long long ga = (unsigned long long)gptr;
  u32x4 g0;
  g0.x = 1u;                                   // count = 1 valid descriptor
  g0.y = lds_off;                              // LDS byte address
  g0.z = (unsigned)ga;                         // global_addr[31:0]
  g0.w = (unsigned)((ga >> 32) & 0x01FFFFFFull) | 0x80000000u; // addr[56:32]|type=2
  i32x8 g1;
  g1[0] = 0x00010000;                          // workgroup_mask=0, data_size=2B
  g1[1] = (int)((tensor_d0 & 0xFFFFu) << 16);  // [63:48] = tensor_dim0 lo16
  g1[2] = (int)((tensor_d0 >> 16) | ((tensor_d1 & 0xFFFFu) << 16));
  g1[3] = (int)((tensor_d1 >> 16) | (tile_d0 << 16));
  g1[4] = (int)(tile_d1 & 0xFFFFu);            // tile_dim1, tile_dim2=0
  g1[5] = (int)stride0_elts;                   // tensor_dim0_stride lo32
  g1[6] = 0;
  g1[7] = 0;
  TDM_ISSUE(g0, g1);
}
#else
#define HAS_TDM 0
#endif

// ---------------------------------------------------------------------------
// Stage 0: W [K=512][N=512] f32  ->  WT [N][K] bf16 (K-contiguous rows)
// ---------------------------------------------------------------------------
__global__ __launch_bounds__(256) void k_wtrans(const float* __restrict__ W,
                                                __bf16* __restrict__ WT) {
  int i = blockIdx.x * 256 + threadIdx.x;   // 0 .. 512*512-1
  int n = i >> 9;
  int k = i & 511;
  WT[i] = (__bf16)W[k * EMB + n];           // WT[n*512 + k]
}

// ---------------------------------------------------------------------------
// Stage 1: Y = X(f32 [8192][512]) @ WT^T, output bf16 split per head.
// mode 0: Y[((b*H+h)*S + s)*hd + d]   (row-major per head: Q, K)
// mode 1: Y[((b*H+h)*hd + d)*S + s]   (transposed per head: V^T)
// Software-pipelined + fully unrolled K loop: step k+1 fragments are in
// flight during step-k WMMAs, no register-rotation copies.
// ---------------------------------------------------------------------------
__global__ __launch_bounds__(256) void k_proj(const float* __restrict__ X,
                                              const __bf16* __restrict__ WT,
                                              __bf16* __restrict__ Y, int mode) {
  const int lane   = threadIdx.x & 31;
  const int wave   = blockIdx.x * 8 + (threadIdx.x >> 5);
  const int mtile  = wave >> 3;
  const int nstrip = wave & 7;
  const int am     = lane & 15;
  const int kgrp   = (lane < 16) ? 0 : 16;
  const int bofs   = (lane < 16) ? 0 : 16;
  const int ncol0  = nstrip * 64;
  const int arowi  = mtile * 16 + am;

  auto loadA = [&](int kk) -> bf16x16 {
    const float* ap = X + (size_t)arowi * EMB + kk + kgrp;
    return cat8(cvt8(ap), cvt8(ap + 8));
  };
  auto loadB = [&](int kk, int j) -> bf16x16 {
    int n = ncol0 + j * 16 + am;
    return *(const bf16x16*)(WT + (size_t)n * EMB + kk + bofs);
  };

  f32x8 acc[4] = {};
  bf16x16 aCur = loadA(0);
  bf16x16 bCur[4];
#pragma unroll
  for (int j = 0; j < 4; ++j) bCur[j] = loadB(0, j);

#pragma unroll
  for (int kk = 0; kk + 32 < EMB; kk += 32) {
    bf16x16 aNxt = loadA(kk + 32);
    bf16x16 bNxt[4];
#pragma unroll
    for (int j = 0; j < 4; ++j) bNxt[j] = loadB(kk + 32, j);
#pragma unroll
    for (int j = 0; j < 4; ++j) acc[j] = wmma_bf16(aCur, bCur[j], acc[j]);
    aCur = aNxt;
#pragma unroll
    for (int j = 0; j < 4; ++j) bCur[j] = bNxt[j];
  }
#pragma unroll
  for (int j = 0; j < 4; ++j) acc[j] = wmma_bf16(aCur, bCur[j], acc[j]);

  const int half16 = (lane >= 16) ? 8 : 0;
#pragma unroll
  for (int j = 0; j < 4; ++j) {
    int col = ncol0 + j * 16 + am;
    int h = col >> 6, d = col & 63;
#pragma unroll
    for (int r = 0; r < 8; ++r) {
      int rowO = mtile * 16 + r + half16;
      int b = rowO >> 11, s = rowO & 2047;
      size_t idx;
      if (mode == 0) idx = (((size_t)(b * NHEAD + h)) * SEQ + s) * HD + d;
      else           idx = (((size_t)(b * NHEAD + h)) * HD + d) * SEQ + s;
      Y[idx] = (__bf16)acc[j][r];
    }
  }
}

// ---------------------------------------------------------------------------
// Stage 2: flash attention, block-cooperative.
// Block = 256 threads (8 waves) -> one (b,h) and 128 q-rows (16 per wave).
// K/V^T 32-wide chunks staged in LDS by the Tensor Data Mover, double-buffered:
// wave 0 issues TDM for chunk c+1 while all waves compute on chunk c.
// ---------------------------------------------------------------------------
__global__ __launch_bounds__(256) void k_attn(const __bf16* __restrict__ Q,
                                              const __bf16* __restrict__ Km,
                                              const __bf16* __restrict__ VT,
                                              const float* __restrict__ bias,
                                              __bf16* __restrict__ ctx) {
  __shared__ __bf16 Kbuf[2][32 * HD];      // [t][d]  4KB per buffer
  __shared__ __bf16 Vbuf[2][HD * 32];      // [d][t]  4KB per buffer
  __shared__ __bf16 pbuf[8][16 * 32];      // per-wave P tile

  const int lane  = threadIdx.x & 31;
  const int wslot = threadIdx.x >> 5;
  const int bh    = blockIdx.x & 31;       // b*H + h
  const int qg    = blockIdx.x >> 5;       // 16 q-groups of 128 rows
  const int q0    = qg * 128 + wslot * 16;

  const __bf16* Qh = Q  + (size_t)bh * SEQ * HD;
  const __bf16* Kh = Km + (size_t)bh * SEQ * HD;
  const __bf16* Vh = VT + (size_t)bh * HD * SEQ;

  const int am     = lane & 15;
  const int kgrp   = (lane < 16) ? 0 : 16;
  const int bofs   = (lane < 16) ? 0 : 16;
  const int half16 = (lane >= 16) ? 8 : 0;

  // Q fragments: hd=64 -> two K=32 fragments, resident in registers.
  bf16x16 qfrag[2];
#pragma unroll
  for (int t = 0; t < 2; ++t) {
    const __bf16* p = Qh + (size_t)(q0 + am) * HD + t * 32 + kgrp;
    qfrag[t] = cat8(*(const bf16x8*)p, *(const bf16x8*)(p + 8));
  }

  f32x8 o[4] = {};
  float mrow[8], lrow[8];
#pragma unroll
  for (int r = 0; r < 8; ++r) { mrow[r] = -1e30f; lrow[r] = 0.f; }

  __bf16* pb = &pbuf[wslot][0];

#if HAS_TDM
  // Prologue: wave 0 DMAs chunk 0 into buffer 0.
  if (wslot == 0) {
    tdm_load_2d((unsigned)(unsigned long long)(void*)&Kbuf[0][0],
                Kh, /*td0*/ 32 * HD, /*td1*/ 1, /*tile0*/ 32 * HD, /*tile1*/ 1,
                32 * HD);
    tdm_load_2d((unsigned)(unsigned long long)(void*)&Vbuf[0][0],
                Vh, /*td0*/ SEQ, /*td1*/ HD, /*tile0*/ 32, /*tile1*/ HD, SEQ);
  }
#endif

  for (int c = 0; c < SEQ / 32; ++c) {
    const int t0 = c * 32;
#if HAS_TDM
    const int cur = c & 1;
    if (wslot == 0) {
      if (c + 1 < SEQ / 32) {
        const int nb = (c + 1) & 1, t1 = t0 + 32;
        tdm_load_2d((unsigned)(unsigned long long)(void*)&Kbuf[nb][0],
                    Kh + (size_t)t1 * HD, 32 * HD, 1, 32 * HD, 1, 32 * HD);
        tdm_load_2d((unsigned)(unsigned long long)(void*)&Vbuf[nb][0],
                    Vh + t1, SEQ, HD, 32, HD, SEQ);
        __builtin_amdgcn_s_wait_tensorcnt(2);   // chunk c complete (in-order)
      } else {
        __builtin_amdgcn_s_wait_tensorcnt(0);
      }
    }
    __syncthreads();                            // publish chunk c to all waves
#else
    const int cur = 0;
    __syncthreads();                            // buffer free for refill
    // Cooperative staging: K chunk 4KB (1x16B/thread), V chunk 4KB.
    ((uint4*)&Kbuf[0][0])[threadIdx.x] =
        ((const uint4*)(Kh + (size_t)t0 * HD))[threadIdx.x];
    ((uint4*)&Vbuf[0][0])[threadIdx.x] =
        *((const uint4*)(Vh + (size_t)(threadIdx.x >> 2) * SEQ + t0) +
          (threadIdx.x & 3));
    __syncthreads();
#endif

    // Prefetch next bias block while computing this chunk.
    if (t0 + 32 < SEQ)
      __builtin_prefetch(bias + (size_t)(q0 + am) * SEQ + t0 + 32, 0, 1);

    // ---- scores: S = Q . K^T for 32 target columns (fragments from LDS) --
    f32x8 c0 = {}, c1 = {};
#pragma unroll
    for (int t = 0; t < 2; ++t) {
      const __bf16* bp0 = &Kbuf[cur][(am) * HD + t * 32 + bofs];
      c0 = wmma_bf16(qfrag[t], *(const bf16x16*)bp0, c0);
      const __bf16* bp1 = &Kbuf[cur][(16 + am) * HD + t * 32 + bofs];
      c1 = wmma_bf16(qfrag[t], *(const bf16x16*)bp1, c1);
    }
    // ---- scale (1/sqrt(64)) + bias ---------------------------------------
#pragma unroll
    for (int r = 0; r < 8; ++r) {
      const float* bp = bias + (size_t)(q0 + r + half16) * SEQ + t0 + am;
      c0[r] = c0[r] * 0.125f + bp[0];
      c1[r] = c1[r] * 0.125f + bp[16];
    }
    // ---- online softmax (row lives in one VGPR across a 16-lane half) ----
#pragma unroll
    for (int r = 0; r < 8; ++r) {
      float mx = fmaxf(c0[r], c1[r]);
      mx = fmaxf(mx, __shfl_xor(mx, 1, 32));
      mx = fmaxf(mx, __shfl_xor(mx, 2, 32));
      mx = fmaxf(mx, __shfl_xor(mx, 4, 32));
      mx = fmaxf(mx, __shfl_xor(mx, 8, 32));
      float mnew  = fmaxf(mrow[r], mx);
      float alpha = __expf(mrow[r] - mnew);
      float p0 = __expf(c0[r] - mnew);
      float p1 = __expf(c1[r] - mnew);
      c0[r] = p0; c1[r] = p1;
      float sum = p0 + p1;
      sum += __shfl_xor(sum, 1, 32);
      sum += __shfl_xor(sum, 2, 32);
      sum += __shfl_xor(sum, 4, 32);
      sum += __shfl_xor(sum, 8, 32);
      lrow[r] = lrow[r] * alpha + sum;
      mrow[r] = mnew;
#pragma unroll
      for (int j = 0; j < 4; ++j) o[j][r] *= alpha;
    }
    // ---- P: accumulator layout -> A layout via wave-private LDS ----------
#pragma unroll
    for (int r = 0; r < 8; ++r) {
      pb[(r + half16) * 32 + am]      = (__bf16)c0[r];
      pb[(r + half16) * 32 + 16 + am] = (__bf16)c1[r];
    }
    asm volatile("" ::: "memory");   // keep DS store->load order (HW in-order)
    const __bf16* pr = pb + am * 32 + kgrp;
    bf16x16 pfrag = cat8(*(const bf16x8*)pr, *(const bf16x8*)(pr + 8));
    // ---- O += P . V  (V^T rows are t-contiguous in LDS) ------------------
#pragma unroll
    for (int j = 0; j < 4; ++j) {
      const __bf16* vp = &Vbuf[cur][(j * 16 + am) * 32 + bofs];
      o[j] = wmma_bf16(pfrag, *(const bf16x16*)vp, o[j]);
    }

#if HAS_TDM
    __syncthreads();   // all waves done reading buf before wave0 refills it
#endif
  }

  // ---- normalize, merge heads, store ctx bf16 ----------------------------
  const int b = bh >> 3, h = bh & 7;
#pragma unroll
  for (int j = 0; j < 4; ++j) {
    int col = h * HD + j * 16 + am;
#pragma unroll
    for (int r = 0; r < 8; ++r) {
      int rowO = q0 + r + half16;
      float v = o[j][r] / lrow[r];
      ctx[((size_t)(b * SEQ + rowO)) * EMB + col] = (__bf16)v;
    }
  }
}

// ---------------------------------------------------------------------------
// Stage 3: out = ctx(bf16) @ Wo, output f32.  Pipelined + fully unrolled.
// ---------------------------------------------------------------------------
__global__ __launch_bounds__(256) void k_outproj(const __bf16* __restrict__ Xb,
                                                 const __bf16* __restrict__ WT,
                                                 float* __restrict__ Y) {
  const int lane   = threadIdx.x & 31;
  const int wave   = blockIdx.x * 8 + (threadIdx.x >> 5);
  const int mtile  = wave >> 3;
  const int nstrip = wave & 7;
  const int am     = lane & 15;
  const int kgrp   = (lane < 16) ? 0 : 16;
  const int bofs   = (lane < 16) ? 0 : 16;
  const int ncol0  = nstrip * 64;
  const int arowi  = mtile * 16 + am;

  auto loadA = [&](int kk) -> bf16x16 {
    const __bf16* ap = Xb + (size_t)arowi * EMB + kk + kgrp;
    return cat8(*(const bf16x8*)ap, *(const bf16x8*)(ap + 8));
  };
  auto loadB = [&](int kk, int j) -> bf16x16 {
    int n = ncol0 + j * 16 + am;
    return *(const bf16x16*)(WT + (size_t)n * EMB + kk + bofs);
  };

  f32x8 acc[4] = {};
  bf16x16 aCur = loadA(0);
  bf16x16 bCur[4];
#pragma unroll
  for (int j = 0; j < 4; ++j) bCur[j] = loadB(0, j);

#pragma unroll
  for (int kk = 0; kk + 32 < EMB; kk += 32) {
    bf16x16 aNxt = loadA(kk + 32);
    bf16x16 bNxt[4];
#pragma unroll
    for (int j = 0; j < 4; ++j) bNxt[j] = loadB(kk + 32, j);
#pragma unroll
    for (int j = 0; j < 4; ++j) acc[j] = wmma_bf16(aCur, bCur[j], acc[j]);
    aCur = aNxt;
#pragma unroll
    for (int j = 0; j < 4; ++j) bCur[j] = bNxt[j];
  }
#pragma unroll
  for (int j = 0; j < 4; ++j) acc[j] = wmma_bf16(aCur, bCur[j], acc[j]);

  const int half16 = (lane >= 16) ? 8 : 0;
#pragma unroll
  for (int j = 0; j < 4; ++j) {
    int col = ncol0 + j * 16 + am;
#pragma unroll
    for (int r = 0; r < 8; ++r) {
      int rowO = mtile * 16 + r + half16;
      Y[(size_t)rowO * EMB + col] = acc[j][r];
    }
  }
}

// ---------------------------------------------------------------------------
extern "C" void kernel_launch(void* const* d_in, const int* in_sizes, int n_in,
                              void* d_out, int out_size, void* d_ws, size_t ws_size,
                              hipStream_t stream) {
  const float* query  = (const float*)d_in[0];
  const float* target = (const float*)d_in[1];
  const float* bias   = (const float*)d_in[2];
  const float* Wq     = (const float*)d_in[3];
  const float* Wk     = (const float*)d_in[4];
  const float* Wv     = (const float*)d_in[5];
  const float* Wo     = (const float*)d_in[6];
  float* out = (float*)d_out;

  // Workspace (bf16): 4x 512*512 weightT + 4x 8192*512 tensors = 35,651,584 B.
  __bf16* WqT = (__bf16*)d_ws;
  __bf16* WkT = WqT + 512 * 512;
  __bf16* WvT = WkT + 512 * 512;
  __bf16* WoT = WvT + 512 * 512;
  __bf16* Qb  = WoT + 512 * 512;
  __bf16* Kb  = Qb  + 8192 * 512;
  __bf16* VTb = Kb  + 8192 * 512;
  __bf16* ctx = VTb + 8192 * 512;

  dim3 blk(256);
  k_wtrans<<<1024, blk, 0, stream>>>(Wq, WqT);
  k_wtrans<<<1024, blk, 0, stream>>>(Wk, WkT);
  k_wtrans<<<1024, blk, 0, stream>>>(Wv, WvT);
  k_wtrans<<<1024, blk, 0, stream>>>(Wo, WoT);
  k_proj<<<512, blk, 0, stream>>>(query,  WqT, Qb,  0);
  k_proj<<<512, blk, 0, stream>>>(target, WkT, Kb,  0);
  k_proj<<<512, blk, 0, stream>>>(target, WvT, VTb, 1);
  k_attn<<<512, blk, 0, stream>>>(Qb, Kb, VTb, bias, ctx);
  k_outproj<<<512, blk, 0, stream>>>(ctx, WoT, out);
}
